// Layer3bit_83356725281467
// MI455X (gfx1250) — matrix operand order repaired and checked by reading, not compiled, not run on hardware
//
#include <hip/hip_runtime.h>
#include <stdint.h>

typedef _Float16 half_t;
typedef __attribute__((ext_vector_type(16))) _Float16 v16h;
typedef __attribute__((ext_vector_type(8)))  _Float16 v8h;
typedef __attribute__((ext_vector_type(4)))  _Float16 v4h;
typedef __attribute__((ext_vector_type(2)))  _Float16 v2h;
typedef __attribute__((ext_vector_type(8)))  float    v8f;
typedef __attribute__((ext_vector_type(4)))  int      v4i;

constexpr int Mdim = 4096, Ndim = 4096, Kdim = 4096;
constexpr int GROUP = 128;

constexpr int BM = 128;   // block tile M
constexpr int BN = 128;   // block tile N
constexpr int BK = 32;    // block tile K (== WMMA K)
constexpr int LDA = 40;   // padded LDS row stride (halves): 80B, 16B-aligned, conflict-free
constexpr int LDB = 40;

__global__ __launch_bounds__(256)
void marlin3bit_wmma_kernel(const half_t* __restrict__ A,
                            const int*    __restrict__ q,
                            const half_t* __restrict__ s,
                            half_t*       __restrict__ C)
{
    __shared__ half_t As[2][BM * LDA];   // A tile, row-major [m][k]
    __shared__ half_t Bt[2][BN * LDB];   // dequantized B tile, transposed [n][k]

    const int tid   = threadIdx.x;
    const int lane  = tid & 31;
    const int wave  = tid >> 5;       // 0..7
    const int waveM = wave & 1;       // 2 waves along M (64 rows each)
    const int waveN = wave >> 1;      // 4 waves along N (32 cols each)

    const int n0 = blockIdx.x * BN;
    const int m0 = blockIdx.y * BM;

    // --- global staging assignments ---
    const int aRow  = tid >> 1;         // 0..127 (32B contiguous per thread)
    const int aSeg  = tid & 1;          // 0..1   (16 halves each)
    const int bCol4 = (tid & 31) * 4;   // 0..124 (4 consecutive n)
    const int bRow4 = (tid >> 5) * 4;   // 0..28  (4 consecutive k)

    v8h aReg0, aReg1;
    v4i qReg[4];
    v4h sReg, msReg;                    // s and -1028*s per column

    v8f acc[4][2] = {};

    const int S = Kdim / BK;            // 128 K-steps

    auto load_regs = [&](int kt) {
        const int k0 = kt * BK;
        const half_t* ap = A + (size_t)(m0 + aRow) * Kdim + k0 + aSeg * 16;
        aReg0 = *(const v8h*)(ap);
        aReg1 = *(const v8h*)(ap + 8);
        const int* qp = q + (size_t)(k0 + bRow4) * Ndim + n0 + bCol4;
        qReg[0] = *(const v4i*)(qp);
        qReg[1] = *(const v4i*)(qp + Ndim);
        qReg[2] = *(const v4i*)(qp + 2 * Ndim);
        qReg[3] = *(const v4i*)(qp + 3 * Ndim);
        sReg  = *(const v4h*)(s + (size_t)(k0 / GROUP) * Ndim + n0 + bCol4);
        msReg = sReg * (_Float16)-1028.0f;
    };

    auto store_lds = [&](int buf) {
        half_t* ad = &As[buf][aRow * LDA + aSeg * 16];
        *(v8h*)(ad)     = aReg0;
        *(v8h*)(ad + 8) = aReg1;
        // dequant 4(k) x 4(n) micro tile:  w*s = (1024+q)*s + (-1028*s)
        // bitcast(0x6400 | q) == 1024 + q exactly for q in [0,7]
#pragma unroll
        for (int c = 0; c < 4; ++c) {
            const v2h sv2 = { sReg[c],  sReg[c]  };
            const v2h ms2 = { msReg[c], msReg[c] };
            union { uint32_t u; v2h h; } u0, u1;
            u0.u = (uint32_t)qReg[0][c] | ((uint32_t)qReg[1][c] << 16) | 0x64006400u;
            u1.u = (uint32_t)qReg[2][c] | ((uint32_t)qReg[3][c] << 16) | 0x64006400u;
            const v2h w01 = u0.h * sv2 + ms2;   // v_pk_fma_f16 (+ v_or3_b32 pack)
            const v2h w23 = u1.h * sv2 + ms2;
            const v4h w = __builtin_shufflevector(w01, w23, 0, 1, 2, 3);
            *(v4h*)(&Bt[buf][(bCol4 + c) * LDB + bRow4]) = w;
        }
    };

    auto load_afrag = [&](int buf, int mi) -> v16h {
        const int row  = waveM * 64 + mi * 16 + (lane & 15);
        const int koff = (lane >> 4) * 8;            // 0 or 8
        const half_t* p = &As[buf][row * LDA + koff];
        const v8h lo = *(const v8h*)(p);             // K = koff .. koff+7
        const v8h hi = *(const v8h*)(p + 16);        // K = koff+16 .. koff+23
        v16h a;
#pragma unroll
        for (int i = 0; i < 8; ++i) { a[i] = lo[i]; a[8 + i] = hi[i]; }
        return a;
    };

    auto load_bfrag = [&](int buf, int ni) -> v16h {
        const int col  = waveN * 32 + ni * 16 + (lane & 15);
        const int koff = (lane >> 4) * 8;            // 0 or 8
        const half_t* p = &Bt[buf][col * LDB + koff];
        const v8h lo = *(const v8h*)(p);             // K = koff .. koff+7
        const v8h hi = *(const v8h*)(p + 16);        // K = koff+16 .. koff+23
        v16h b;
#pragma unroll
        for (int i = 0; i < 8; ++i) { b[i] = lo[i]; b[8 + i] = hi[i]; }
        return b;
    };

    // --- prologue ---
    load_regs(0);
    store_lds(0);
    __syncthreads();

    // --- main loop: double-buffered; global loads + dequant overlap WMMA ---
    for (int kt = 0; kt < S; ++kt) {
        const int cur = kt & 1;
        if (kt + 1 < S) load_regs(kt + 1);

        // hoist ALL fragment loads so one dscnt wait covers the batch and the
        // 8 independent WMMAs issue back-to-back
        v16h bfrag[2], afrag[4];
        bfrag[0] = load_bfrag(cur, 0);
        bfrag[1] = load_bfrag(cur, 1);
#pragma unroll
        for (int mi = 0; mi < 4; ++mi) afrag[mi] = load_afrag(cur, mi);

#pragma unroll
        for (int mi = 0; mi < 4; ++mi) {
            acc[mi][0] = __builtin_amdgcn_wmma_f32_16x16x32_f16(
                false, afrag[mi], false, bfrag[0], (short)0, acc[mi][0], false, false);
            acc[mi][1] = __builtin_amdgcn_wmma_f32_16x16x32_f16(
                false, afrag[mi], false, bfrag[1], (short)0, acc[mi][1], false, false);
        }

        if (kt + 1 < S) store_lds(1 - cur);
        __syncthreads();
    }

    // --- epilogue: C layout: VGPR r -> M = r + (lane>=16 ? 8 : 0), N = lane&15 ---
#pragma unroll
    for (int mi = 0; mi < 4; ++mi)
#pragma unroll
        for (int ni = 0; ni < 2; ++ni) {
            const int col = n0 + waveN * 32 + ni * 16 + (lane & 15);
            const int rowBase = m0 + waveM * 64 + mi * 16 + ((lane >> 4) * 8);
#pragma unroll
            for (int r = 0; r < 8; ++r)
                C[(size_t)(rowBase + r) * Ndim + col] = (half_t)acc[mi][ni][r];
        }
}

extern "C" void kernel_launch(void* const* d_in, const int* in_sizes, int n_in,
                              void* d_out, int out_size, void* d_ws, size_t ws_size,
                              hipStream_t stream) {
    (void)in_sizes; (void)n_in; (void)out_size; (void)d_ws; (void)ws_size;
    const half_t* A = (const half_t*)d_in[0];
    const int*    q = (const int*)d_in[1];
    const half_t* s = (const half_t*)d_in[2];
    half_t*       C = (half_t*)d_out;

    dim3 grid(Ndim / BN, Mdim / BM);   // 32 x 32 blocks
    marlin3bit_wmma_kernel<<<grid, 256, 0, stream>>>(A, q, s, C);
}
